// RelationLayer_4518305595806
// MI455X (gfx1250) — matrix operand order repaired and checked by reading, not compiled, not run on hardware
//
#include <hip/hip_runtime.h>

// Problem constants (match reference)
#define L_DIM 64
#define B_DIM 32
#define D_IN  512
#define D_OUT 512
#define D_MLP 128
#define C0    256
#define C1    128
#define C_TOT 384
#define NROW  2048   // L*B
#define K0TOT 768    // 2*C_TOT (w0 K extent)

typedef __attribute__((ext_vector_type(16))) _Float16 v16h;
typedef __attribute__((ext_vector_type(8)))  float    v8f;

// ---------------- static device scratch (deterministic; fully rewritten each launch) ----
__device__ _Float16 g_xh  [NROW * D_IN];        // masked x, f16
__device__ _Float16 g_cw0h[C0 * D_IN];          // conv_w0 f16 (channel-major: [n][k])
__device__ _Float16 g_cw1h[C1 * 3 * D_IN];      // conv_w1 f16 (channel-major: [n][k])
__device__ _Float16 g_w0t [D_MLP * K0TOT];      // w0^T  f16: [n][k], k in [0,768)
__device__ _Float16 g_w1t [D_MLP * D_MLP];      // w1^T  f16: [n][k]
__device__ _Float16 g_w2t [D_MLP * D_MLP];
__device__ _Float16 g_w3t [D_MLP * D_MLP];
__device__ _Float16 g_w4t [D_MLP * D_MLP];
__device__ _Float16 g_w5t [D_OUT * D_MLP];      // w5^T  f16: [n][k], n in [0,512)
__device__ float    g_y   [NROW * C_TOT];       // conv out, pre-BN
__device__ float    g_mean[C_TOT];
__device__ float    g_istd[C_TOT];
__device__ _Float16 g_h16 [NROW * C_TOT];       // post-BN features, f16
__device__ float    g_a   [NROW * D_MLP];       // h @ w0[:C]
__device__ float    g_bb  [NROW * D_MLP];       // h @ w0[C:]
__device__ float    g_pmean[NROW * D_MLP];      // masked mean over j (atomic accum)

// ---------------- helpers -------------------------------------------------------------
__device__ __forceinline__ float lrelu(float x) { return x >= 0.0f ? x : 0.1f * x; }

__device__ __forceinline__ v8f splat8(float v) {
  v8f c;
#pragma unroll
  for (int r = 0; r < 8; ++r) c[r] = v;
  return c;
}

__device__ __forceinline__ v8f wmma_f16(v16h a, v16h b, v8f c) {
  return __builtin_amdgcn_wmma_f32_16x16x32_f16(false, a, false, b, (short)0, c,
                                                false, false);
}

// A fragment (16x32 f16, row-major src, ld = row stride).
// lanes 0-15: row=lane, K {0..7,16..23}; lanes 16-31: row=lane-16, K {8..15,24..31}
__device__ __forceinline__ v16h load_A16(const _Float16* base, int ld) {
  const int lane = threadIdx.x & 31;
  const int row  = lane & 15;
  const int kb   = (lane & 16) >> 1;   // 0 or 8
  const _Float16* p = base + row * ld + kb;
  v16h a;
#pragma unroll
  for (int j = 0; j < 8; ++j) a[j] = p[j];
#pragma unroll
  for (int j = 0; j < 8; ++j) a[8 + j] = p[16 + j];
  return a;
}

// A fragment from f32 source (converted on load)
__device__ __forceinline__ v16h load_A16f(const float* base, int ld) {
  const int lane = threadIdx.x & 31;
  const int row  = lane & 15;
  const int kb   = (lane & 16) >> 1;
  const float* p = base + row * ld + kb;
  v16h a;
#pragma unroll
  for (int j = 0; j < 8; ++j) a[j] = (_Float16)p[j];
#pragma unroll
  for (int j = 0; j < 8; ++j) a[8 + j] = (_Float16)p[16 + j];
  return a;
}

// B fragment (32x16) from N-major storage: element(k,n) = base[n*ld + k].
// Two 16B runs per lane.
// lanes 0-15: col=lane, K {0..7,16..23}; lanes 16-31: col=lane-16, K {8..15,24..31}
__device__ __forceinline__ v16h load_B16T(const _Float16* base, int ld) {
  const int lane = threadIdx.x & 31;
  const int col  = lane & 15;
  const int kb   = (lane & 16) >> 1;   // 0 or 8
  const _Float16* p = base + col * ld + kb;
  v16h b;
#pragma unroll
  for (int j = 0; j < 8; ++j) b[j] = p[j];
#pragma unroll
  for (int j = 0; j < 8; ++j) b[8 + j] = p[16 + j];
  return b;
}

// D fragment store (16x16 f32). lanes 0-15: col=lane, rows 0..7; lanes 16-31: rows 8..15
__device__ __forceinline__ void store_D_f32(float* base, int ld, v8f d) {
  const int lane = threadIdx.x & 31;
  const int col  = lane & 15;
  const int rb   = (lane & 16) >> 1;
#pragma unroll
  for (int r = 0; r < 8; ++r) base[(rb + r) * ld + col] = d[r];
}

__device__ __forceinline__ void store_D_leaky_f16(_Float16* base, int ld, v8f d) {
  const int lane = threadIdx.x & 31;
  const int col  = lane & 15;
  const int rb   = (lane & 16) >> 1;
#pragma unroll
  for (int r = 0; r < 8; ++r) base[(rb + r) * ld + col] = (_Float16)lrelu(d[r]);
}

// ---------------- kernels -------------------------------------------------------------

// Convert inputs/weights to f16 (transposing the row-major MLP weights to N-major),
// apply mask to x, zero pmean accumulator.
__global__ void k_prep(const float* x, const float* xs_mask, const float* cw0,
                       const float* cw1, const float* w0, const float* w1,
                       const float* w2, const float* w3, const float* w4,
                       const float* w5) {
  const int tid = blockIdx.x * blockDim.x + threadIdx.x;
  const int np  = gridDim.x * blockDim.x;
  for (int i = tid; i < NROW * D_IN; i += np)
    g_xh[i] = (_Float16)(x[i] * xs_mask[i >> 9]);
  for (int i = tid; i < C0 * D_IN; i += np)      g_cw0h[i] = (_Float16)cw0[i];
  for (int i = tid; i < C1 * 3 * D_IN; i += np)  g_cw1h[i] = (_Float16)cw1[i];
  // w0 (768 x 128) -> g_w0t[n*768 + k]
  for (int i = tid; i < K0TOT * D_MLP; i += np) {
    const int k = i >> 7, n = i & (D_MLP - 1);
    g_w0t[n * K0TOT + k] = (_Float16)w0[i];
  }
  // w1..w4 (128 x 128) -> [n*128 + k]
  for (int i = tid; i < D_MLP * D_MLP; i += np) {
    const int k = i >> 7, n = i & (D_MLP - 1);
    const int t = n * D_MLP + k;
    g_w1t[t] = (_Float16)w1[i];
    g_w2t[t] = (_Float16)w2[i];
    g_w3t[t] = (_Float16)w3[i];
    g_w4t[t] = (_Float16)w4[i];
  }
  // w5 (128 x 512) -> g_w5t[n*128 + k]
  for (int i = tid; i < D_MLP * D_OUT; i += np) {
    const int k = i / D_OUT, n = i % D_OUT;
    g_w5t[n * D_MLP + k] = (_Float16)w5[i];
  }
  for (int i = tid; i < NROW * D_MLP; i += np)   g_pmean[i] = 0.0f;
}

// conv window=1: y[:, 0:256] = x @ cw0^T + cb0.  grid(128,4) x 128 thr
__global__ void __launch_bounds__(128) k_conv0(const float* cb0) {
  const int mt   = blockIdx.x;               // 16-row tile of (l,b) rows
  const int wave = threadIdx.x >> 5;
  const int nt   = blockIdx.y * 4 + wave;    // 0..15 -> channels nt*16
  const int col  = threadIdx.x & 15;
  v8f acc = splat8(cb0[nt * 16 + col]);
  const _Float16* A = g_xh + mt * 16 * D_IN;
  const _Float16* B = g_cw0h + nt * 16 * D_IN;
#pragma unroll 4
  for (int k = 0; k < D_IN; k += 32)
    acc = wmma_f16(load_A16(A + k, D_IN), load_B16T(B + k, D_IN), acc);
  store_D_f32(g_y + mt * 16 * C_TOT + nt * 16, C_TOT, acc);
}

// conv window=3: y[:, 256:384] over [x[l-1];x[l];x[l+1]].  grid(128,2) x 128 thr
__global__ void __launch_bounds__(128) k_conv1(const float* cb1) {
  const int mt   = blockIdx.x;
  const int l    = mt >> 1;                  // B=32 -> 2 tiles per l
  const int wave = threadIdx.x >> 5;
  const int nt   = blockIdx.y * 4 + wave;    // 0..7 -> channels 256+nt*16
  const int col  = threadIdx.x & 15;
  v8f acc = splat8(cb1[nt * 16 + col]);
#pragma unroll
  for (int w = 0; w < 3; ++w) {
    const int ls = l + w - 1;
    if (ls < 0 || ls >= L_DIM) continue;     // zero padding (uniform per block)
    const _Float16* A = g_xh + (ls * B_DIM + (mt & 1) * 16) * D_IN;
    const _Float16* B = g_cw1h + nt * 16 * (3 * D_IN) + w * D_IN;
    for (int k = 0; k < D_IN; k += 32)
      acc = wmma_f16(load_A16(A + k, D_IN), load_B16T(B + k, 3 * D_IN), acc);
  }
  store_D_f32(g_y + mt * 16 * C_TOT + C0 + nt * 16, C_TOT, acc);
}

// per-channel batch stats (biased var) over 2048 samples
__global__ void k_bnstats() {
  const int c = blockIdx.x;
  float s = 0.0f, q = 0.0f;
  for (int r = threadIdx.x; r < NROW; r += blockDim.x) {
    const float v = g_y[r * C_TOT + c];
    s += v; q += v * v;
  }
  __shared__ float sh_s[256], sh_q[256];
  sh_s[threadIdx.x] = s; sh_q[threadIdx.x] = q;
  __syncthreads();
  for (int o = 128; o > 0; o >>= 1) {
    if (threadIdx.x < o) {
      sh_s[threadIdx.x] += sh_s[threadIdx.x + o];
      sh_q[threadIdx.x] += sh_q[threadIdx.x + o];
    }
    __syncthreads();
  }
  if (threadIdx.x == 0) {
    const float m   = sh_s[0] * (1.0f / NROW);
    const float var = sh_q[0] * (1.0f / NROW) - m * m;
    g_mean[c] = m;
    g_istd[c] = rsqrtf(var + 1e-5f);
  }
}

// normalize + leaky + mask -> h16
__global__ void k_bnapply(const float* g0, const float* bt0, const float* g1,
                          const float* bt1, const float* xs_mask) {
  const int tid = blockIdx.x * blockDim.x + threadIdx.x;
  const int np  = gridDim.x * blockDim.x;
  for (int i = tid; i < NROW * C_TOT; i += np) {
    const int c = i % C_TOT;
    const int r = i / C_TOT;
    const float gm = (c < C0) ? g0[c]  : g1[c - C0];
    const float bt = (c < C0) ? bt0[c] : bt1[c - C0];
    float v = (g_y[i] - g_mean[c]) * g_istd[c] * gm + bt;
    g_h16[i] = (_Float16)(lrelu(v) * xs_mask[r]);
  }
}

// a = h @ w0[:C], bb = h @ w0[C:]  (K = 384).  grid(128,4) x 128 thr
__global__ void __launch_bounds__(128) k_ab() {
  const int mt   = blockIdx.x;
  const int wave = threadIdx.x >> 5;
  const int nt   = blockIdx.y * 4 + wave;    // 0..15
  const int ncol = (nt & 7) * 16;
  float* outp    = (nt < 8) ? g_a : g_bb;
  const _Float16* Bb = g_w0t + ncol * K0TOT + ((nt < 8) ? 0 : C_TOT);
  v8f acc = splat8(0.0f);
#pragma unroll 4
  for (int ks = 0; ks < 12; ++ks)
    acc = wmma_f16(load_A16(g_h16 + mt * 16 * C_TOT + ks * 32, C_TOT),
                   load_B16T(Bb + ks * 32, K0TOT), acc);
  store_D_f32(outp + mt * 16 * D_MLP + ncol, D_MLP, acc);
}

// Fused pairwise MLP + mean over j.  grid(i=64, jc=8) x 128 thr (4 waves).
// Each wave owns 32 output cols; w1/w2/w3 B-fragments live in registers.
__global__ void __launch_bounds__(128) k_pair(const float* b0, const float* b1,
                                              const float* b2, const float* b3,
                                              const float* xs_mask) {
  __shared__ float    s_a [B_DIM * D_MLP];   // a_i tile (f32)
  __shared__ float    s_b0[D_MLP];
  __shared__ _Float16 s_p0[B_DIM * D_MLP];
  __shared__ _Float16 s_p1[B_DIM * D_MLP];
  __shared__ _Float16 s_p2[B_DIM * D_MLP];

  const int i    = blockIdx.x;
  const int tid  = threadIdx.x;
  const int wave = tid >> 5;
  const int lane = tid & 31;
  const int col  = lane & 15;
  const int nbase = wave * 32;

  for (int idx = tid; idx < B_DIM * D_MLP; idx += 128)
    s_a[idx] = g_a[i * B_DIM * D_MLP + idx];
  if (tid < D_MLP) s_b0[tid] = b0[tid];

  // weight fragments in registers: [n-tile][k-step], contiguous loads from N-major
  v16h w1f[2][4], w2f[2][4], w3f[2][4];
#pragma unroll
  for (int n = 0; n < 2; ++n)
#pragma unroll
    for (int k = 0; k < 4; ++k) {
      const int off = (nbase + n * 16) * D_MLP + k * 32;
      w1f[n][k] = load_B16T(g_w1t + off, D_MLP);
      w2f[n][k] = load_B16T(g_w2t + off, D_MLP);
      w3f[n][k] = load_B16T(g_w3t + off, D_MLP);
    }
  v8f bias1[2], bias2[2], bias3[2];
#pragma unroll
  for (int n = 0; n < 2; ++n) {
    bias1[n] = splat8(b1[nbase + n * 16 + col]);
    bias2[n] = splat8(b2[nbase + n * 16 + col]);
    bias3[n] = splat8(b3[nbase + n * 16 + col]);
  }
  v8f sum[2][2];
#pragma unroll
  for (int m = 0; m < 2; ++m)
#pragma unroll
    for (int n = 0; n < 2; ++n) sum[m][n] = splat8(0.0f);

  __syncthreads();

  for (int jj = 0; jj < 8; ++jj) {
    const int j = blockIdx.y * 8 + jj;
    const float* bbr = g_bb + j * B_DIM * D_MLP;
    // p0 = leaky(a_i + bb_j + b0)
    for (int idx = tid; idx < B_DIM * D_MLP; idx += 128) {
      const float v = s_a[idx] + bbr[idx] + s_b0[idx & (D_MLP - 1)];
      s_p0[idx] = (_Float16)lrelu(v);
    }
    __syncthreads();

    // layer 1: p1 = leaky(p0 @ w1 + b1)
#pragma unroll
    for (int m = 0; m < 2; ++m) {
      v8f a0 = bias1[0], a1 = bias1[1];
#pragma unroll
      for (int k = 0; k < 4; ++k) {
        const v16h af = load_A16(s_p0 + m * 16 * D_MLP + k * 32, D_MLP);
        a0 = wmma_f16(af, w1f[0][k], a0);
        a1 = wmma_f16(af, w1f[1][k], a1);
      }
      store_D_leaky_f16(s_p1 + m * 16 * D_MLP + nbase,      D_MLP, a0);
      store_D_leaky_f16(s_p1 + m * 16 * D_MLP + nbase + 16, D_MLP, a1);
    }
    __syncthreads();

    // layer 2: p2 = leaky(p1 @ w2 + b2)
#pragma unroll
    for (int m = 0; m < 2; ++m) {
      v8f a0 = bias2[0], a1 = bias2[1];
#pragma unroll
      for (int k = 0; k < 4; ++k) {
        const v16h af = load_A16(s_p1 + m * 16 * D_MLP + k * 32, D_MLP);
        a0 = wmma_f16(af, w2f[0][k], a0);
        a1 = wmma_f16(af, w2f[1][k], a1);
      }
      store_D_leaky_f16(s_p2 + m * 16 * D_MLP + nbase,      D_MLP, a0);
      store_D_leaky_f16(s_p2 + m * 16 * D_MLP + nbase + 16, D_MLP, a1);
    }
    __syncthreads();

    // layer 3: sum += leaky(p2 @ w3 + b3)
#pragma unroll
    for (int m = 0; m < 2; ++m) {
      v8f a0 = bias3[0], a1 = bias3[1];
#pragma unroll
      for (int k = 0; k < 4; ++k) {
        const v16h af = load_A16(s_p2 + m * 16 * D_MLP + k * 32, D_MLP);
        a0 = wmma_f16(af, w3f[0][k], a0);
        a1 = wmma_f16(af, w3f[1][k], a1);
      }
#pragma unroll
      for (int r = 0; r < 8; ++r) {
        sum[m][0][r] += lrelu(a0[r]);
        sum[m][1][r] += lrelu(a1[r]);
      }
    }
    __syncthreads();   // protect s_p0 for next iteration
  }

  // masked mean contribution (j split over gridDim.y blocks)
  const int rb = (lane & 16) >> 1;
  const float inv = 1.0f / (float)L_DIM;
#pragma unroll
  for (int m = 0; m < 2; ++m)
#pragma unroll
    for (int n = 0; n < 2; ++n)
#pragma unroll
      for (int r = 0; r < 8; ++r) {
        const int row = i * B_DIM + m * 16 + rb + r;
        const float v = sum[m][n][r] * inv * xs_mask[row];
        atomicAdd(&g_pmean[row * D_MLP + nbase + n * 16 + col], v);
      }
}

// out = leaky(leaky(pmean @ w4 + b4) @ w5 + b5) * mask.  grid 128 x 128 thr
__global__ void __launch_bounds__(128) k_final(const float* b4, const float* b5,
                                               const float* xs_mask, float* out) {
  __shared__ _Float16 s_t1[16 * D_MLP];
  const int mt   = blockIdx.x;          // 16-row tile
  const int tid  = threadIdx.x;
  const int wave = tid >> 5;
  const int lane = tid & 31;
  const int col  = lane & 15;
  const int rb   = (lane & 16) >> 1;

  {  // t1 = leaky(pmean @ w4 + b4), N = 128 (wave covers 32 cols)
    const int n0 = wave * 32;
    v8f a0 = splat8(b4[n0 + col]);
    v8f a1 = splat8(b4[n0 + 16 + col]);
#pragma unroll
    for (int k = 0; k < 4; ++k) {
      const v16h af = load_A16f(g_pmean + mt * 16 * D_MLP + k * 32, D_MLP);
      a0 = wmma_f16(af, load_B16T(g_w4t + n0 * D_MLP + k * 32,        D_MLP), a0);
      a1 = wmma_f16(af, load_B16T(g_w4t + (n0 + 16) * D_MLP + k * 32, D_MLP), a1);
    }
    store_D_leaky_f16(s_t1 + n0,      D_MLP, a0);
    store_D_leaky_f16(s_t1 + n0 + 16, D_MLP, a1);
  }
  __syncthreads();

  // out tile: N = 512 (wave covers 128 cols)
  const int cb = wave * 128;
  float mk[8];
#pragma unroll
  for (int r = 0; r < 8; ++r) mk[r] = xs_mask[mt * 16 + rb + r];
#pragma unroll
  for (int n = 0; n < 8; ++n) {
    const int nc = cb + n * 16;
    v8f acc = splat8(b5[nc + col]);
#pragma unroll
    for (int k = 0; k < 4; ++k)
      acc = wmma_f16(load_A16(s_t1 + k * 32, D_MLP),
                     load_B16T(g_w5t + nc * D_MLP + k * 32, D_MLP), acc);
#pragma unroll
    for (int r = 0; r < 8; ++r)
      out[(mt * 16 + rb + r) * D_OUT + nc + col] = lrelu(acc[r]) * mk[r];
  }
}

// ---------------- launch --------------------------------------------------------------
extern "C" void kernel_launch(void* const* d_in, const int* in_sizes, int n_in,
                              void* d_out, int out_size, void* d_ws, size_t ws_size,
                              hipStream_t stream) {
  const float* x    = (const float*)d_in[0];
  const float* mask = (const float*)d_in[1];
  const float* cw0  = (const float*)d_in[2];
  const float* cb0  = (const float*)d_in[3];
  const float* g0   = (const float*)d_in[4];
  const float* bt0  = (const float*)d_in[5];
  const float* cw1  = (const float*)d_in[6];
  const float* cb1  = (const float*)d_in[7];
  const float* g1   = (const float*)d_in[8];
  const float* bt1  = (const float*)d_in[9];
  const float* w0   = (const float*)d_in[10];
  const float* b0   = (const float*)d_in[11];
  const float* w1   = (const float*)d_in[12];
  const float* b1   = (const float*)d_in[13];
  const float* w2   = (const float*)d_in[14];
  const float* b2   = (const float*)d_in[15];
  const float* w3   = (const float*)d_in[16];
  const float* b3   = (const float*)d_in[17];
  const float* w4   = (const float*)d_in[18];
  const float* b4   = (const float*)d_in[19];
  const float* w5   = (const float*)d_in[20];
  const float* b5   = (const float*)d_in[21];
  float* out = (float*)d_out;

  k_prep<<<256, 256, 0, stream>>>(x, mask, cw0, cw1, w0, w1, w2, w3, w4, w5);
  k_conv0<<<dim3(128, 4), 128, 0, stream>>>(cb0);
  k_conv1<<<dim3(128, 2), 128, 0, stream>>>(cb1);
  k_bnstats<<<C_TOT, 256, 0, stream>>>();
  k_bnapply<<<1024, 256, 0, stream>>>(g0, bt0, g1, bt1, mask);
  k_ab<<<dim3(128, 4), 128, 0, stream>>>();
  k_pair<<<dim3(64, 8), 128, 0, stream>>>(b0, b1, b2, b3, mask);
  k_final<<<128, 128, 0, stream>>>(b4, b5, mask, out);
}
